// FastKANLayer_28733331210511
// MI455X (gfx1250) — matrix operand order repaired and checked by reading, not compiled, not run on hardware
//
#include <hip/hip_runtime.h>

typedef __attribute__((ext_vector_type(16))) _Float16 v16h;
typedef __attribute__((ext_vector_type(8)))  _Float16 h8;
typedef __attribute__((ext_vector_type(8)))  float    v8f;

union V16 { v16h v; _Float16 h[16]; };
union PK2 { _Float16 h[2]; unsigned u; };

#define NB   8
#define NI   16
#define NJ   16
#define NS   256
#define NH   64
#define NC   67
#define KD   512     // mid * d = 8 * 64
#define AKS  264     // padded LDS stride for ak rows (bank-conflict-free)

// ---------- K1: per-(b,i,s) row stats of x_in and proj ----------
__global__ __launch_bounds__(256) void k_rowstats(
    const float* __restrict__ x_in, const float* __restrict__ proj,
    float* __restrict__ xe, float* __restrict__ am,
    float* __restrict__ pm, float* __restrict__ pm2) {
  int b = blockIdx.x >> 4, i = blockIdx.x & 15;
  int s = threadIdx.x;
  size_t base = (((size_t)b * NI + i) * NS + s) * NH;
  float su = 0.f, asu = 0.f, ps = 0.f, ps2 = 0.f;
  for (int f = 0; f < NH; f += 4) {
    float4 v = *(const float4*)(x_in + base + f);
    su += v.x + v.y + v.z + v.w;
    asu += fabsf(v.x) + fabsf(v.y) + fabsf(v.z) + fabsf(v.w);
    float4 p = *(const float4*)(proj + base + f);
    ps += p.x + p.y + p.z + p.w;
    ps2 += p.x * p.x + p.y * p.y + p.z * p.z + p.w * p.w;
  }
  int o = (b * NI + i) * NS + s;
  xe[o] = su * (1.f / 64.f);
  am[o] = asu * (1.f / 64.f);
  pm[o] = ps * (1.f / 64.f);
  pm2[o] = ps2 * (1.f / 64.f);
}

// ---------- K2: x_norm ----------
__global__ __launch_bounds__(256) void k_xnorm(const float* __restrict__ xe,
                                               float* __restrict__ xn) {
  int row0 = blockIdx.x * NS;
  int t = threadIdx.x;
  __shared__ float rb[256];
  float e = xe[row0 + t];
  rb[t] = fabsf(e);
  __syncthreads();
  for (int off = 128; off >= 1; off >>= 1) {
    if (t < off) rb[t] = fmaxf(rb[t], rb[t + off]);
    __syncthreads();
  }
  float mx = rb[0];
  float v = e / (mx + 1e-8f);
  xn[row0 + t] = fminf(fmaxf(v, -0.99f), 0.99f);
}

// ---------- K3: spline edge response + edge energies ----------
__global__ __launch_bounds__(256) void k_edge(
    const float* __restrict__ xnorm, const float* __restrict__ absmean,
    const float* __restrict__ sw_g, const float* __restrict__ base_scales,
    const float* __restrict__ omiga, float* __restrict__ ero,
    float* __restrict__ eE) {
  int b = blockIdx.x >> 4, i = blockIdx.x & 15;
  int t = threadIdx.x;  // = s
  __shared__ float s_sw[NJ * NC];
  __shared__ float s_e[NJ];
  for (int p = t; p < NJ * NC; p += 256) s_sw[p] = sw_g[i * NJ * NC + p];
  if (t < NJ) s_e[t] = 0.f;
  __syncthreads();
  int row = (b * NI + i) * NS + t;
  float xv = xnorm[row];
  float amv = absmean[row];
  float sil = xv / (1.f + __expf(-xv));
  const float h = 2.0f / 66.0f;
  float acc[NJ];
#pragma unroll
  for (int jj = 0; jj < NJ; ++jj) acc[jj] = 0.f;
  for (int c = 0; c < NC; ++c) {
    float g = -1.f + h * (float)c;
    float d = fabsf(xv - g) / h;
    float bas;
    if (d < 1.f) bas = 2.f / 3.f - d * d + d * d * d * 0.5f;
    else if (d < 2.f) { float e2 = 2.f - d; bas = e2 * e2 * e2 * (1.f / 6.f); }
    else bas = 0.f;
#pragma unroll
    for (int jj = 0; jj < NJ; ++jj) acc[jj] += bas * s_sw[jj * NC + c];
  }
#pragma unroll
  for (int jj = 0; jj < NJ; ++jj) {
    float er = sil * base_scales[i * NJ + jj] + acc[jj];
    float eo = er + fabsf(omiga[i * NJ + jj]);
    ero[(((size_t)(b * NI + i) * NJ + jj) * NS) + t] = eo;
    atomicAdd(&s_e[jj], fabsf(eo) * amv);
  }
  __syncthreads();
  if (t < NJ) eE[(b * NI + i) * NJ + t] = s_e[t] * (1.f / 256.f);
}

// ---------- K4: mask + wp coefficient ----------
__global__ __launch_bounds__(256) void k_mask(
    const float* __restrict__ eE, const float* __restrict__ tau,
    const float* __restrict__ temp, float* __restrict__ mask,
    float* __restrict__ cwp) {
  int idx = blockIdx.x * 256 + threadIdx.x;  // 2048
  int ij = idx & 255;                        // i*16+j
  float e = eE[idx];
  float ta = fabsf(tau[ij]);
  float tval = fabsf(temp[0]) + 1e-4f;
  float m = 1.f / (1.f + __expf(-(e - ta) / tval));
  mask[idx] = m;
  cwp[idx] = (e / (ta + 1e-8f)) * m;
}

// ---------- K5: algebraic LN stats for K and Q rows ----------
__global__ __launch_bounds__(256) void k_lnstats(
    const float* __restrict__ cwp, const float* __restrict__ pm,
    const float* __restrict__ pm2, float* __restrict__ kmu,
    float* __restrict__ krs, float* __restrict__ qmu,
    float* __restrict__ qrs) {
  int b = blockIdx.x >> 4, j = blockIdx.x & 15;
  int s = threadIdx.x;
  float km = 0.f, k2 = 0.f, qm = 0.f, q2 = 0.f;
#pragma unroll
  for (int i = 0; i < 8; ++i) {
    float ck = cwp[(b * NI + i) * NJ + j];
    float cq = cwp[(b * NI + 8 + i) * NJ + j];
    float pk = pm[(b * NI + i) * NS + s], pk2 = pm2[(b * NI + i) * NS + s];
    float pq = pm[(b * NI + 8 + i) * NS + s], pq2 = pm2[(b * NI + 8 + i) * NS + s];
    km += ck * pk;  k2 += ck * ck * pk2;
    qm += cq * pq;  q2 += cq * cq * pq2;
  }
  km *= 0.125f; k2 *= 0.125f; qm *= 0.125f; q2 *= 0.125f;
  int o = (b * NJ + j) * NS + s;
  kmu[o] = km; krs[o] = rsqrtf(fmaxf(k2 - km * km, 0.f) + 1e-5f);
  qmu[o] = qm; qrs[o] = rsqrtf(fmaxf(q2 - qm * qm, 0.f) + 1e-5f);
}

// ---------- K5b: materialize layernormed K/Q in f16 ----------
// Kn/Qn layout: [b*16+j][s][k], k = i*64 + f (i in 0..7; Q uses input i+8)
__global__ __launch_bounds__(256) void k_kqbuild(
    const float* __restrict__ proj, const float* __restrict__ cwp,
    const float* __restrict__ kmu, const float* __restrict__ krs,
    const float* __restrict__ qmu, const float* __restrict__ qrs,
    _Float16* __restrict__ Kn, _Float16* __restrict__ Qn) {
  int b = blockIdx.x >> 4, j = blockIdx.x & 15;
  int t = threadIdx.x;
  int bj = b * NJ + j;
  int k = t * 2;            // this thread's element pair within a row
  int i = k >> 6, f = k & 63;
  float cwk = cwp[(b * NI + i) * NJ + j];
  float cwq = cwp[(b * NI + 8 + i) * NJ + j];
  const float* srcK = proj + (((size_t)b * NI + i) * NS) * NH + f;
  const float* srcQ = proj + (((size_t)b * NI + 8 + i) * NS) * NH + f;
  for (int s = 0; s < NS; ++s) {
    float km = kmu[bj * NS + s], kr = krs[bj * NS + s];
    float qm = qmu[bj * NS + s], qr = qrs[bj * NS + s];
    float2 vk = *(const float2*)(srcK + (size_t)s * NH);
    float2 vq = *(const float2*)(srcQ + (size_t)s * NH);
    PK2 pk, pq;
    pk.h[0] = (_Float16)((vk.x * cwk - km) * kr);
    pk.h[1] = (_Float16)((vk.y * cwk - km) * kr);
    pq.h[0] = (_Float16)((vq.x * cwq - qm) * qr);
    pq.h[1] = (_Float16)((vq.y * cwq - qm) * qr);
    size_t rbase = ((size_t)bj * NS + s) * KD + k;
    *(unsigned*)(Kn + rbase) = pk.u;
    *(unsigned*)(Qn + rbase) = pq.u;
  }
}

// ---------- K6: masked mean over i + affine layernorm -> xn (f16) ----------
__global__ __launch_bounds__(256) void k_xmean(
    const float* __restrict__ ero, const float* __restrict__ mask,
    const float* __restrict__ x_in, const float* __restrict__ lnw,
    const float* __restrict__ lnb, float* __restrict__ xres,
    _Float16* __restrict__ xnh) {
  int b = blockIdx.x >> 4, j = blockIdx.x & 15;
  int t = threadIdx.x;
  __shared__ float s_md[NI];
  __shared__ float rb[256];
  if (t < NI) s_md[t] = mask[(b * NI + t) * NJ + j] * (1.f / 16.f);
  __syncthreads();
  int sl = t >> 6, f = t & 63;
  for (int it = 0; it < NS / 4; ++it) {
    int s = it * 4 + sl;
    float acc = 0.f;
#pragma unroll
    for (int i = 0; i < NI; ++i) {
      float er = ero[(((size_t)(b * NI + i) * NJ + j) * NS) + s];
      acc += er * s_md[i] * x_in[(((size_t)(b * NI + i) * NS + s) * NH) + f];
    }
    size_t idxo = (((size_t)(b * NJ + j) * NS + s) * NH) + f;
    xres[idxo] = acc;
    rb[t] = acc;
    __syncthreads();
    for (int off = 32; off >= 1; off >>= 1) {
      if ((t & 63) < off) rb[t] += rb[t + off];
      __syncthreads();
    }
    float mu = rb[t & ~63] * (1.f / 64.f);
    __syncthreads();
    float d = acc - mu;
    rb[t] = d * d;
    __syncthreads();
    for (int off = 32; off >= 1; off >>= 1) {
      if ((t & 63) < off) rb[t] += rb[t + off];
      __syncthreads();
    }
    float var = rb[t & ~63] * (1.f / 64.f);
    float rs = rsqrtf(var + 1e-5f);
    float xv = d * rs * lnw[j * NH + f] + lnb[j * NH + f];
    xnh[idxo] = (_Float16)xv;
    __syncthreads();
  }
}

// ---------- K7: transpose W2 -> f16 [j][o][f] ----------
__global__ __launch_bounds__(256) void k_w2t(const float* __restrict__ W2,
                                             _Float16* __restrict__ w2t) {
  int j = blockIdx.x;
  for (int p = threadIdx.x; p < NH * NH; p += 256) {
    int f = p >> 6, o = p & 63;
    w2t[((size_t)j * NH + o) * NH + f] = (_Float16)W2[((size_t)j * NH + f) * NH + o];
  }
}

// ---------- K8: x_prime = xn @ W2[j] + b  (WMMA f16) ----------
__global__ __launch_bounds__(256) void k_xprime(
    const _Float16* __restrict__ xnh, const _Float16* __restrict__ w2t,
    const float* __restrict__ bparam, float* __restrict__ xp_out,
    _Float16* __restrict__ xpT) {
  int b = blockIdx.x >> 4, j = blockIdx.x & 15;
  int t = threadIdx.x, lane = t & 31, w = t >> 5;
  int lm = lane & 15;
  bool hi = lane >= 16;
  int bj = b * NJ + j;
  for (int tile = w; tile < 64; tile += 8) {
    int s0 = (tile >> 2) * 16, o0 = (tile & 3) * 16;
    v8f acc = {};
    const _Float16* arow = xnh + ((size_t)bj * NS + (s0 + lm)) * NH;
    const _Float16* brow = w2t + ((size_t)j * NH + (o0 + lm)) * NH;
#pragma unroll
    for (int ks = 0; ks < 2; ++ks) {
      int k0 = ks * 32;
      V16 ua, ub;
      int abase = k0 + (hi ? 8 : 0);
      *(h8*)&ua.h[0] = *(const h8*)(arow + abase);
      *(h8*)&ua.h[8] = *(const h8*)(arow + abase + 16);
      int bbase = k0 + (hi ? 8 : 0);
      *(h8*)&ub.h[0] = *(const h8*)(brow + bbase);
      *(h8*)&ub.h[8] = *(const h8*)(brow + bbase + 16);
      acc = __builtin_amdgcn_wmma_f32_16x16x32_f16(false, ua.v, false, ub.v,
                                                   (short)0, acc, false, false);
    }
#pragma unroll
    for (int r = 0; r < 8; ++r) {
      int M = r + (hi ? 8 : 0);
      int s = s0 + M, o = o0 + lm;
      float val = acc[r] + bparam[((size_t)j * NS + s) * NH + o];
      xp_out[((size_t)bj * NS + s) * NH + o] = val;
      xpT[((size_t)bj * NH + o) * NS + s] = (_Float16)val;
    }
  }
}

// ---------- K9: depthwise 3x3 conv on 16x16 view ----------
__global__ __launch_bounds__(256) void k_conv(const float* __restrict__ xp,
                                              const float* __restrict__ cw,
                                              float* __restrict__ w3) {
  int b = blockIdx.x >> 4, j = blockIdx.x & 15;
  int s = threadIdx.x;
  int r = s >> 4, c = s & 15;
  size_t bj = (size_t)(b * NJ + j);
  const float* base = xp + bj * NS * NH;
  for (int o = 0; o < NH; ++o) {
    float sum = 0.f;
#pragma unroll
    for (int kr = 0; kr < 3; ++kr) {
      int rr = r + kr - 1;
      if (rr < 0 || rr > 15) continue;
#pragma unroll
      for (int kc = 0; kc < 3; ++kc) {
        int cc = c + kc - 1;
        if (cc < 0 || cc > 15) continue;
        sum += base[(size_t)(rr * 16 + cc) * NH + o] *
               cw[(j * NH + o) * 9 + kr * 3 + kc];
      }
    }
    w3[bj * NS * NH + (size_t)s * NH + o] = sum;
  }
}

// ---------- K10: fused attention + softmax + glob + final out ----------
__global__ __launch_bounds__(256) void k_attn(
    const _Float16* __restrict__ Kn, const _Float16* __restrict__ Qn,
    const _Float16* __restrict__ xpT, const float* __restrict__ w3,
    const float* __restrict__ xres, const float* __restrict__ alpha,
    const float* __restrict__ beta, const float* __restrict__ gamma_p,
    const float* __restrict__ theta, const float* __restrict__ temp,
    float* __restrict__ out) {
  int blk = blockIdx.x;
  int b = blk >> 8;
  int j = (blk >> 4) & 15;
  int s0 = (blk & 15) << 4;
  int t = threadIdx.x, lane = t & 31, w = t >> 5;
  int lm = lane & 15;
  bool hi = lane >= 16;
  int bj = b * NJ + j;

  __shared__ float s_raw[16][256];
  __shared__ _Float16 s_ak[16 * AKS];

  float tval = fabsf(temp[0]) + 1e-4f;
  float invscale = 1.0f / (sqrtf(512.0f) * tval);

  // Phase A: raw[16, 0:256] = K @ Q^T (K dim = 512), pure f16 loads + WMMA
  int srow = s0 + lm;
  const _Float16* Krow = Kn + ((size_t)bj * NS + srow) * KD;
  __builtin_prefetch(Krow, 0, 0);  // global_prefetch_b8: warm K panel
  int t0a = (2 * w) * 16, t0b = t0a + 16;
  const _Float16* Qrow_a = Qn + ((size_t)bj * NS + (t0a + lm)) * KD;
  const _Float16* Qrow_b = Qn + ((size_t)bj * NS + (t0b + lm)) * KD;
  v8f acc0 = {}, acc1 = {};
  for (int ks = 0; ks < 16; ++ks) {
    int k0 = ks * 32;
    V16 ua, ub0, ub1;
    int abase = k0 + (hi ? 8 : 0);
    *(h8*)&ua.h[0] = *(const h8*)(Krow + abase);
    *(h8*)&ua.h[8] = *(const h8*)(Krow + abase + 16);
    int bbase = k0 + (hi ? 8 : 0);
    *(h8*)&ub0.h[0] = *(const h8*)(Qrow_a + bbase);
    *(h8*)&ub0.h[8] = *(const h8*)(Qrow_a + bbase + 16);
    *(h8*)&ub1.h[0] = *(const h8*)(Qrow_b + bbase);
    *(h8*)&ub1.h[8] = *(const h8*)(Qrow_b + bbase + 16);
    acc0 = __builtin_amdgcn_wmma_f32_16x16x32_f16(false, ua.v, false, ub0.v,
                                                  (short)0, acc0, false, false);
    acc1 = __builtin_amdgcn_wmma_f32_16x16x32_f16(false, ua.v, false, ub1.v,
                                                  (short)0, acc1, false, false);
  }
#pragma unroll
  for (int r = 0; r < 8; ++r) {
    int M = r + (hi ? 8 : 0);
    s_raw[M][t0a + lm] = acc0[r] * invscale;
    s_raw[M][t0b + lm] = acc1[r] * invscale;
  }
  __syncthreads();

  // Phase B: softmax rows + build ak = |beta|*W1 + |alpha|*I in f16
  float beta_a = fabsf(beta[j]);
  float alpha_a = fabsf(alpha[j]);
#pragma unroll
  for (int rr = 0; rr < 2; ++rr) {
    int row = 2 * w + rr;
    float v[8];
    float m = -1e30f;
#pragma unroll
    for (int q = 0; q < 8; ++q) {
      v[q] = s_raw[row][lane + 32 * q];
      m = fmaxf(m, v[q]);
    }
#pragma unroll
    for (int off = 16; off >= 1; off >>= 1) m = fmaxf(m, __shfl_xor(m, off, 32));
    float sum = 0.f;
#pragma unroll
    for (int q = 0; q < 8; ++q) {
      v[q] = __expf(v[q] - m);
      sum += v[q];
    }
#pragma unroll
    for (int off = 16; off >= 1; off >>= 1) sum += __shfl_xor(sum, off, 32);
    float inv = 1.0f / sum;
#pragma unroll
    for (int q = 0; q < 8; ++q) {
      int tt = lane + 32 * q;
      float a = beta_a * v[q] * inv;
      if (tt == s0 + row) a += alpha_a;
      s_ak[row * AKS + tt] = (_Float16)a;
    }
  }
  __syncthreads();

  // Phase C/D: glob = ak @ x_prime, residuals, write out
  if (w < 4) {
    int o0 = w * 16;
    v8f g = {};
    const _Float16* xpTb = xpT + ((size_t)bj * NH + o0 + lm) * NS;
    const _Float16* akrow = &s_ak[lm * AKS];
    for (int ks = 0; ks < 8; ++ks) {
      int k0 = ks * 32;
      V16 ua, ub;
      int abase = k0 + (hi ? 8 : 0);
      *(h8*)&ua.h[0] = *(const h8*)(akrow + abase);
      *(h8*)&ua.h[8] = *(const h8*)(akrow + abase + 16);
      int bbase = k0 + (hi ? 8 : 0);
      *(h8*)&ub.h[0] = *(const h8*)(xpTb + bbase);
      *(h8*)&ub.h[8] = *(const h8*)(xpTb + bbase + 16);
      g = __builtin_amdgcn_wmma_f32_16x16x32_f16(false, ua.v, false, ub.v,
                                                 (short)0, g, false, false);
    }
    float th = fabsf(theta[j]);
    float gm = gamma_p[j];
#pragma unroll
    for (int r = 0; r < 8; ++r) {
      int M = r + (hi ? 8 : 0);
      size_t idx = ((size_t)bj * NS + (s0 + M)) * NH + (o0 + lm);
      out[idx] = g[r] + th * w3[idx] + gm * xres[idx];
    }
  }
}

// ---------- launch ----------
extern "C" void kernel_launch(void* const* d_in, const int* in_sizes, int n_in,
                              void* d_out, int out_size, void* d_ws,
                              size_t ws_size, hipStream_t stream) {
  (void)in_sizes; (void)n_in; (void)out_size; (void)ws_size;
  const float* x_in   = (const float*)d_in[0];
  const float* proj   = (const float*)d_in[1];
  const float* sw     = (const float*)d_in[2];
  const float* bs     = (const float*)d_in[3];
  const float* tau    = (const float*)d_in[4];
  const float* temp   = (const float*)d_in[5];
  const float* om     = (const float*)d_in[6];
  const float* W2     = (const float*)d_in[7];
  const float* bparam = (const float*)d_in[8];
  const float* lnw    = (const float*)d_in[9];
  const float* lnb    = (const float*)d_in[10];
  const float* alpha  = (const float*)d_in[11];
  const float* beta   = (const float*)d_in[12];
  const float* gam    = (const float*)d_in[13];
  const float* theta  = (const float*)d_in[14];
  const float* convw  = (const float*)d_in[15];

  float* out = (float*)d_out;
  float* xprime_out = out + (size_t)NB * NJ * NS * NH;

  char* wsp = (char*)d_ws;
  size_t off = 0;
  auto alloc = [&](size_t bytes) -> void* {
    void* p = wsp + off;
    off = (off + bytes + 255) & ~(size_t)255;
    return p;
  };
  const size_t NBIS = (size_t)NB * NI * NS;        // 32768
  const size_t NBJS = (size_t)NB * NJ * NS;        // 32768
  const size_t NBJSH = NBJS * NH;                  // 2097152
  float*    d_xe  = (float*)alloc(NBIS * 4);
  float*    d_am  = (float*)alloc(NBIS * 4);
  float*    d_xn  = (float*)alloc(NBIS * 4);
  float*    d_pm  = (float*)alloc(NBIS * 4);
  float*    d_pm2 = (float*)alloc(NBIS * 4);
  float*    d_ero = (float*)alloc((size_t)NB * NI * NJ * NS * 4);
  float*    d_eE  = (float*)alloc((size_t)NB * NI * NJ * 4);
  float*    d_msk = (float*)alloc((size_t)NB * NI * NJ * 4);
  float*    d_cwp = (float*)alloc((size_t)NB * NI * NJ * 4);
  float*    d_kmu = (float*)alloc(NBJS * 4);
  float*    d_krs = (float*)alloc(NBJS * 4);
  float*    d_qmu = (float*)alloc(NBJS * 4);
  float*    d_qrs = (float*)alloc(NBJS * 4);
  _Float16* d_Kn  = (_Float16*)alloc(NBJS * KD * 2);   // 33.5 MB
  _Float16* d_Qn  = (_Float16*)alloc(NBJS * KD * 2);   // 33.5 MB
  _Float16* d_xnh = (_Float16*)alloc(NBJSH * 2);
  float*    d_xrs = (float*)alloc(NBJSH * 4);
  _Float16* d_xpT = (_Float16*)alloc(NBJSH * 2);
  float*    d_w3  = (float*)alloc(NBJSH * 4);
  _Float16* d_w2t = (_Float16*)alloc((size_t)NJ * NH * NH * 2);

  k_rowstats<<<NB * NI, 256, 0, stream>>>(x_in, proj, d_xe, d_am, d_pm, d_pm2);
  k_xnorm<<<NB * NI, 256, 0, stream>>>(d_xe, d_xn);
  k_edge<<<NB * NI, 256, 0, stream>>>(d_xn, d_am, sw, bs, om, d_ero, d_eE);
  k_mask<<<(NB * NI * NJ) / 256, 256, 0, stream>>>(d_eE, tau, temp, d_msk, d_cwp);
  k_lnstats<<<NB * NJ, 256, 0, stream>>>(d_cwp, d_pm, d_pm2, d_kmu, d_krs,
                                         d_qmu, d_qrs);
  k_kqbuild<<<NB * NJ, 256, 0, stream>>>(proj, d_cwp, d_kmu, d_krs, d_qmu,
                                         d_qrs, d_Kn, d_Qn);
  k_xmean<<<NB * NJ, 256, 0, stream>>>(d_ero, d_msk, x_in, lnw, lnb, d_xrs, d_xnh);
  k_w2t<<<NJ, 256, 0, stream>>>(W2, d_w2t);
  k_xprime<<<NB * NJ, 256, 0, stream>>>(d_xnh, d_w2t, bparam, xprime_out, d_xpT);
  k_conv<<<NB * NJ, 256, 0, stream>>>(xprime_out, convw, d_w3);
  k_attn<<<NB * NJ * (NS / 16), 256, 0, stream>>>(
      d_Kn, d_Qn, d_xpT, d_w3, d_xrs, alpha, beta, gam, theta, temp, out);
}
